// DA2QN_45148696215824
// MI455X (gfx1250) — matrix operand order repaired and checked by reading, not compiled, not run on hardware
//
#include <hip/hip_runtime.h>
#include <hip/hip_bf16.h>

// ---------------------------------------------------------------------------
// CDNA5 / gfx1250 WMMA implementation of the DA2QN forward pass.
// All GEMM-heavy stages use v_wmma_f32_16x16x32_f16 (f16 in, f32 accumulate).
// ---------------------------------------------------------------------------

typedef __attribute__((ext_vector_type(16))) _Float16 v16h;
typedef __attribute__((ext_vector_type(8)))  float    v8f;

#define DEV __device__ __forceinline__

DEV v8f wmma_f16(v16h a, v16h b, v8f c) {
  // 8 args: (neg_a, A, neg_b, B, c_mod, C, reuse_a, reuse_b)
  return __builtin_amdgcn_wmma_f32_16x16x32_f16(false, a, false, b, (short)0, c,
                                                false, false);
}

// --- Fragment loaders (layouts per CDNA5 ISA 7.12.2, wave32) ---------------
// A (16x32 f16, row-major MxK source; src pre-offset to (m0,k0)):
//   lanes 0-15: M=lane,    K = {0..7} U {16..23}
//   lanes 16-31: M=lane-16, K = {8..15} U {24..31}
DEV v16h load_a_rm(const _Float16* src, int ld, int lane) {
  const int m  = lane & 15;
  const int kb = (lane >> 4) << 3;
  const _Float16* p = src + m * ld + kb;
  v16h a;
#pragma unroll
  for (int h = 0; h < 16; ++h) a[h] = p[(h < 8) ? h : h + 8];
  return a;
}

// B (32x16 f16) from row-major KxN source; src pre-offset to (k0,n0):
//   lane = N; lanes 0-15 hold K={0..7}U{16..23}, lanes 16-31 hold K={8..15}U{24..31}.
DEV v16h load_b_kn(const _Float16* src, int ld, int lane) {
  const int n  = lane & 15;
  const int kb = (lane >> 4) << 3;
  const _Float16* p = src + kb * ld + n;
  v16h b;
#pragma unroll
  for (int h = 0; h < 16; ++h) b[h] = p[((h < 8) ? h : h + 8) * ld];
  return b;
}

// B (32x16 f16) from row-major NxK source (weight matrices [out][in]);
// src pre-offset to (n0 row, k0 col). K = kb+{0..7} U kb+{16..23} per lane.
DEV v16h load_b_nk(const _Float16* src, int ld, int lane) {
  const int n  = lane & 15;
  const int kb = (lane >> 4) << 3;
  const _Float16* p = src + n * ld + kb;
  v16h b;
#pragma unroll
  for (int h = 0; h < 16; ++h) b[h] = p[(h < 8) ? h : h + 8];
  return b;
}

DEV int imin(int a, int b) { return a < b ? a : b; }

// ---------------------------------------------------------------------------
// f32 -> f16 conversion helpers (one-time per launch)
// ---------------------------------------------------------------------------
__global__ void DA2QN_cvt(_Float16* __restrict__ dst,
                          const float* __restrict__ src, int n) {
  int i = blockIdx.x * 256 + threadIdx.x;
  if (i < n) dst[i] = (_Float16)src[i];
}

// wfc5 [18][520] -> f16 padded to [32][520] (rows 18..31 zero)
__global__ void DA2QN_padw5(_Float16* __restrict__ dst,
                            const float* __restrict__ src) {
  int i = blockIdx.x * 256 + threadIdx.x;
  if (i < 32 * 520) {
    int r = i / 520, c = i - r * 520;
    dst[i] = (r < 18) ? (_Float16)src[r * 520 + c] : (_Float16)0.f;
  }
}

// ---------------------------------------------------------------------------
// conv1: frames[B,4,84,84] f16, w1h[32][256], stride 4, 8x8 -> x1h [B,32,400]
// Implicit GEMM: M = 400 positions/batch (25 tiles), N = 32, K = 256.
// One wave per 16x32 output tile.
// ---------------------------------------------------------------------------
__global__ __launch_bounds__(32) void DA2QN_conv1(
    const _Float16* __restrict__ fr, const _Float16* __restrict__ w,
    const float* __restrict__ bias, _Float16* __restrict__ out) {
  const int wid = blockIdx.x;
  const int b = wid / 25, mt = wid - b * 25;
  const int m0 = mt * 16;
  const int lane = threadIdx.x;
  const int lm = lane & 15, hi = lane >> 4;
  const int m = m0 + lm;
  const int oy = m / 20, ox = m - oy * 20;
  const _Float16* in = fr + (size_t)b * (4 * 84 * 84) + (oy * 4) * 84 + ox * 4;
  const int kb = hi * 8;

  v8f acc0 = {}, acc1 = {};
#pragma unroll 2
  for (int k0 = 0; k0 < 256; k0 += 32) {
    if (k0 + 32 < 256) __builtin_prefetch(w + lm * 256 + k0 + 32, 0, 1);
    v16h a;
#pragma unroll
    for (int h = 0; h < 16; ++h) {
      const int k = k0 + kb + ((h < 8) ? h : h + 8);
      const int c = k >> 6, ry = (k >> 3) & 7, rx = k & 7;
      a[h] = in[(c * 84 + ry) * 84 + rx];
    }
    v16h b0 = load_b_nk(w + k0, 256, lane);
    v16h b1 = load_b_nk(w + 16 * 256 + k0, 256, lane);
    acc0 = wmma_f16(a, b0, acc0);
    acc1 = wmma_f16(a, b1, acc1);
  }
#pragma unroll
  for (int r = 0; r < 8; ++r) {
    const int mm = m0 + r + hi * 8;
    float v0 = acc0[r] + bias[lm];
    float v1 = acc1[r] + bias[16 + lm];
    v0 = v0 > 0.f ? v0 : 0.f;
    v1 = v1 > 0.f ? v1 : 0.f;
    out[((size_t)b * 32 + lm) * 400 + mm] = (_Float16)v0;
    out[((size_t)b * 32 + 16 + lm) * 400 + mm] = (_Float16)v1;
  }
}

// ---------------------------------------------------------------------------
// conv2: x1h[B,32,400](20x20), w2h[64][512], stride 2, 4x4 -> x2h [B,64,81]
// M = 81/batch (6 tiles, last partial), N = 64, K = 512.
// ---------------------------------------------------------------------------
__global__ __launch_bounds__(32) void DA2QN_conv2(
    const _Float16* __restrict__ x1, const _Float16* __restrict__ w,
    const float* __restrict__ bias, _Float16* __restrict__ out) {
  const int wid = blockIdx.x;
  const int b = wid / 6, mt = wid - b * 6;
  const int m0 = mt * 16;
  const int lane = threadIdx.x;
  const int lm = lane & 15, hi = lane >> 4;
  const int mc = imin(m0 + lm, 80);
  const int oy = mc / 9, ox = mc - oy * 9;
  const _Float16* in = x1 + (size_t)b * (32 * 400) + (oy * 2) * 20 + ox * 2;
  const int kb = hi * 8;

  v8f acc[4];
#pragma unroll
  for (int t = 0; t < 4; ++t) { v8f z = {}; acc[t] = z; }

#pragma unroll 1
  for (int k0 = 0; k0 < 512; k0 += 32) {
    v16h a;
#pragma unroll
    for (int h = 0; h < 16; ++h) {
      const int k = k0 + kb + ((h < 8) ? h : h + 8);
      const int c = k >> 4, ry = (k >> 2) & 3, rx = k & 3;
      a[h] = in[c * 400 + ry * 20 + rx];
    }
#pragma unroll
    for (int t = 0; t < 4; ++t) {
      v16h bb = load_b_nk(w + (size_t)(t * 16) * 512 + k0, 512, lane);
      acc[t] = wmma_f16(a, bb, acc[t]);
    }
  }
#pragma unroll
  for (int r = 0; r < 8; ++r) {
    const int mm = m0 + r + hi * 8;
    if (mm < 81) {
#pragma unroll
      for (int t = 0; t < 4; ++t) {
        const int n = t * 16 + lm;
        float v = acc[t][r] + bias[n];
        v = v > 0.f ? v : 0.f;
        out[((size_t)b * 64 + n) * 81 + mm] = (_Float16)v;
      }
    }
  }
}

// ---------------------------------------------------------------------------
// conv3: x2h[B,64,81](9x9), w3h[64][576], stride 1, 3x3 -> fmh [B][64 pos][64 ch]
// (positions padded 49->64 with zeros for the attention GEMMs; ReLU applied)
// M = 49/batch (4 tiles), N = 64, K = 576.
// ---------------------------------------------------------------------------
__global__ __launch_bounds__(32) void DA2QN_conv3(
    const _Float16* __restrict__ x2, const _Float16* __restrict__ w,
    const float* __restrict__ bias, _Float16* __restrict__ fmh) {
  const int wid = blockIdx.x;
  const int b = wid >> 2, mt = wid & 3;
  const int m0 = mt * 16;
  const int lane = threadIdx.x;
  const int lm = lane & 15, hi = lane >> 4;
  const int mc = imin(m0 + lm, 48);
  const int oy = mc / 7, ox = mc - oy * 7;
  const _Float16* in = x2 + (size_t)b * (64 * 81) + oy * 9 + ox;
  const int kb = hi * 8;

  v8f acc[4];
#pragma unroll
  for (int t = 0; t < 4; ++t) { v8f z = {}; acc[t] = z; }

#pragma unroll 1
  for (int k0 = 0; k0 < 576; k0 += 32) {
    v16h a;
#pragma unroll
    for (int h = 0; h < 16; ++h) {
      const int k = k0 + kb + ((h < 8) ? h : h + 8);
      const int c = k / 9, rem = k - c * 9;
      const int ry = rem / 3, rx = rem - ry * 3;
      a[h] = in[c * 81 + ry * 9 + rx];
    }
#pragma unroll
    for (int t = 0; t < 4; ++t) {
      v16h bb = load_b_nk(w + (size_t)(t * 16) * 576 + k0, 576, lane);
      acc[t] = wmma_f16(a, bb, acc[t]);
    }
  }
#pragma unroll
  for (int r = 0; r < 8; ++r) {
    const int mm = m0 + r + hi * 8;  // 0..63
#pragma unroll
    for (int t = 0; t < 4; ++t) {
      const int n = t * 16 + lm;
      float v = acc[t][r] + bias[n];
      v = v > 0.f ? v : 0.f;
      fmh[(size_t)b * 4096 + mm * 64 + n] = (mm < 49) ? (_Float16)v
                                                      : (_Float16)0.f;
    }
  }
}

// ---------------------------------------------------------------------------
// Attention block: one workgroup (4 waves) per batch image, all in LDS.
// theta/phi/g 1x1 convs, 49x49 softmax attention, wp, theta2, goal attention,
// aggregation -> agg[B,64] f16. All GEMMs padded to 64 and done with WMMA.
// ---------------------------------------------------------------------------
__global__ __launch_bounds__(128) void DA2QN_attn(
    const _Float16* __restrict__ fmh, const float* __restrict__ goals,
    const _Float16* __restrict__ wt1, const float* __restrict__ bt1,
    const _Float16* __restrict__ wphi, const float* __restrict__ bphi,
    const _Float16* __restrict__ wg, const float* __restrict__ bg,
    const _Float16* __restrict__ wp, const float* __restrict__ bp,
    const _Float16* __restrict__ wt2, const float* __restrict__ bt2,
    const float* __restrict__ walpha, const float* __restrict__ balpha,
    _Float16* __restrict__ agg) {
  __shared__ _Float16 FM[64 * 64];   // fm   [pos][ch]
  __shared__ _Float16 TH[64 * 32];   // theta1 [pos][ch]
  __shared__ _Float16 PHT[32 * 64];  // phi^T  [ch][pos]
  __shared__ _Float16 G[64 * 32];    // g      [pos][ch]
  __shared__ float    SIMf[64 * 64]; // sim (f32 for softmax)
  __shared__ _Float16 SCO[64 * 64];  // softmax(sim)
  __shared__ _Float16 SA1[64 * 32];  // scores @ g
  __shared__ _Float16 SAh[64 * 64];  // sa = SA1 @ wp^T + bp   [pos][ch]
  __shared__ _Float16 TH2[64 * 32];  // theta2 [pos][ch]
  __shared__ float    ALPHA[32];
  __shared__ float    ATT[64];

  const int b = blockIdx.x;
  const int tid = threadIdx.x;
  const int lane = tid & 31;
  const int wv = tid >> 5;
  const int lm = lane & 15, hi = lane >> 4;

  // Stage fm tile into LDS.
  const _Float16* fsrc = fmh + (size_t)b * 4096;
  for (int i = tid; i < 4096; i += 128) FM[i] = fsrc[i];
  __syncthreads();

  // theta1 / phi / g : [64x64pad] @ [64x32], K=64. 24 wave-tiles.
  for (int t = wv; t < 24; t += 4) {
    const int which = t >> 3;             // 0=theta, 1=phi, 2=g
    const int mt = (t >> 1) & 3, nt = t & 1;
    const int m0 = mt * 16, n0 = nt * 16;
    const _Float16* W = (which == 0) ? wt1 : (which == 1) ? wphi : wg;
    const float* Bv = (which == 0) ? bt1 : (which == 1) ? bphi : bg;
    v8f acc = {};
    for (int k0 = 0; k0 < 64; k0 += 32) {
      v16h a = load_a_rm(FM + m0 * 64 + k0, 64, lane);
      v16h bb = load_b_nk(W + n0 * 64 + k0, 64, lane);
      acc = wmma_f16(a, bb, acc);
    }
    const int n = n0 + lm;
    const float bias = Bv[n];
#pragma unroll
    for (int r = 0; r < 8; ++r) {
      const int m = m0 + r + hi * 8;
      const _Float16 v = (_Float16)(acc[r] + bias);
      if (which == 0) TH[m * 32 + n] = v;
      else if (which == 1) PHT[n * 64 + m] = v;   // store transposed
      else G[m * 32 + n] = v;
    }
  }
  __syncthreads();

  // sim = theta1 @ phi^T : 64x64, K=32. 16 wave-tiles, one WMMA each.
  for (int t = wv; t < 16; t += 4) {
    const int mt = t >> 2, nt = t & 3;
    const int m0 = mt * 16, n0 = nt * 16;
    v16h a = load_a_rm(TH + m0 * 32, 32, lane);
    v16h bb = load_b_kn(PHT + n0, 64, lane);
    v8f acc = {};
    acc = wmma_f16(a, bb, acc);
    const int n = n0 + lm;
#pragma unroll
    for (int r = 0; r < 8; ++r) SIMf[(m0 + r + hi * 8) * 64 + n] = acc[r];
  }
  __syncthreads();

  // Row softmax over the 49 valid columns; zero the padding.
  if (tid < 64) {
    if (tid < 49) {
      const float* row = SIMf + tid * 64;
      float mx = row[0];
      for (int j = 1; j < 49; ++j) mx = fmaxf(mx, row[j]);
      float s = 0.f;
      for (int j = 0; j < 49; ++j) s += __expf(row[j] - mx);
      const float inv = 1.f / s;
      for (int j = 0; j < 49; ++j)
        SCO[tid * 64 + j] = (_Float16)(__expf(row[j] - mx) * inv);
      for (int j = 49; j < 64; ++j) SCO[tid * 64 + j] = (_Float16)0.f;
    } else {
      for (int j = 0; j < 64; ++j) SCO[tid * 64 + j] = (_Float16)0.f;
    }
  }
  __syncthreads();

  // SA1 = scores @ g : 64x32, K=64. 8 wave-tiles.
  for (int t = wv; t < 8; t += 4) {
    const int mt = t >> 1, nt = t & 1;
    const int m0 = mt * 16, n0 = nt * 16;
    v8f acc = {};
    for (int k0 = 0; k0 < 64; k0 += 32) {
      v16h a = load_a_rm(SCO + m0 * 64 + k0, 64, lane);
      v16h bb = load_b_kn(G + k0 * 32 + n0, 32, lane);
      acc = wmma_f16(a, bb, acc);
    }
    const int n = n0 + lm;
#pragma unroll
    for (int r = 0; r < 8; ++r)
      SA1[(m0 + r + hi * 8) * 32 + n] = (_Float16)acc[r];
  }
  __syncthreads();

  // SA = SA1 @ wp^T + bp : 64x64, K=32. 16 wave-tiles.
  for (int t = wv; t < 16; t += 4) {
    const int mt = t >> 2, nt = t & 3;
    const int m0 = mt * 16, n0 = nt * 16;
    v16h a = load_a_rm(SA1 + m0 * 32, 32, lane);
    v16h bb = load_b_nk(wp + n0 * 32, 32, lane);
    v8f acc = {};
    acc = wmma_f16(a, bb, acc);
    const int n = n0 + lm;
    const float bias = bp[n];
#pragma unroll
    for (int r = 0; r < 8; ++r)
      SAh[(m0 + r + hi * 8) * 64 + n] = (_Float16)(acc[r] + bias);
  }
  __syncthreads();

  // alpha_g = goals @ walpha^T + balpha   (tiny; one lane per channel)
  if (tid < 32) {
    float s = balpha[tid];
    const float* gl = goals + (size_t)b * 8;
#pragma unroll
    for (int j = 0; j < 8; ++j) s += gl[j] * walpha[tid * 8 + j];
    ALPHA[tid] = s;
  }
  // theta2 = SA @ wt2^T + bt2 : 64x32, K=64. 8 wave-tiles.
  for (int t = wv; t < 8; t += 4) {
    const int mt = t >> 1, nt = t & 1;
    const int m0 = mt * 16, n0 = nt * 16;
    v8f acc = {};
    for (int k0 = 0; k0 < 64; k0 += 32) {
      v16h a = load_a_rm(SAh + m0 * 64 + k0, 64, lane);
      v16h bb = load_b_nk(wt2 + n0 * 64 + k0, 64, lane);
      acc = wmma_f16(a, bb, acc);
    }
    const int n = n0 + lm;
    const float bias = bt2[n];
#pragma unroll
    for (int r = 0; r < 8; ++r)
      TH2[(m0 + r + hi * 8) * 32 + n] = (_Float16)(acc[r] + bias);
  }
  __syncthreads();

  // att[n] = alpha . theta2[n,:]
  if (tid < 49) {
    float s = 0.f;
    for (int c = 0; c < 32; ++c) s += ALPHA[c] * (float)TH2[tid * 32 + c];
    ATT[tid] = s;
  }
  __syncthreads();
  if (tid == 0) {
    float mx = ATT[0];
    for (int j = 1; j < 49; ++j) mx = fmaxf(mx, ATT[j]);
    float s = 0.f;
    for (int j = 0; j < 49; ++j) { float e = __expf(ATT[j] - mx); ATT[j] = e; s += e; }
    const float inv = 1.f / s;
    for (int j = 0; j < 49; ++j) ATT[j] *= inv;
  }
  __syncthreads();

  // agg[c] = sum_n att[n] * sa[n][c]
  if (tid < 64) {
    float s = 0.f;
    for (int m = 0; m < 49; ++m) s += ATT[m] * (float)SAh[m * 64 + tid];
    agg[(size_t)b * 64 + tid] = (_Float16)s;
  }
}

// ---------------------------------------------------------------------------
// fc4: h = relu(agg[1024,64] @ wfc4^T[64,512] + b) -> hh [1024][512] f16
// ---------------------------------------------------------------------------
__global__ __launch_bounds__(32) void DA2QN_fc4(
    const _Float16* __restrict__ agg, const _Float16* __restrict__ w,
    const float* __restrict__ bias, _Float16* __restrict__ hh) {
  const int wid = blockIdx.x;             // 64 mtiles * 8 ngroups
  const int mt = wid >> 3, ng = wid & 7;
  const int m0 = mt * 16, n0 = ng * 64;
  const int lane = threadIdx.x;
  const int lm = lane & 15, hi = lane >> 4;

  v8f acc[4];
#pragma unroll
  for (int t = 0; t < 4; ++t) { v8f z = {}; acc[t] = z; }

  for (int k0 = 0; k0 < 64; k0 += 32) {
    v16h a = load_a_rm(agg + (size_t)m0 * 64 + k0, 64, lane);
#pragma unroll
    for (int t = 0; t < 4; ++t) {
      v16h bb = load_b_nk(w + (size_t)(n0 + t * 16) * 64 + k0, 64, lane);
      acc[t] = wmma_f16(a, bb, acc[t]);
    }
  }
#pragma unroll
  for (int t = 0; t < 4; ++t) {
    const int n = n0 + t * 16 + lm;
    const float bias_n = bias[n];
#pragma unroll
    for (int r = 0; r < 8; ++r) {
      const int m = m0 + r + hi * 8;
      float v = acc[t][r] + bias_n;
      v = v > 0.f ? v : 0.f;
      hh[(size_t)m * 512 + n] = (_Float16)v;
    }
  }
}

// ---------------------------------------------------------------------------
// fc5: out = [h || goals] @ wfc5^T + b.  M=1024, N=18 (padded to 32), K=520.
// The concat is fused into the A-fragment gather.
// ---------------------------------------------------------------------------
__global__ __launch_bounds__(32) void DA2QN_fc5(
    const _Float16* __restrict__ hh, const float* __restrict__ goals,
    const _Float16* __restrict__ w,  // padded [32][520]
    const float* __restrict__ bias, float* __restrict__ out) {
  const int m0 = blockIdx.x * 16;
  const int lane = threadIdx.x;
  const int lm = lane & 15, hi = lane >> 4;
  const int m = m0 + lm;
  const int kbA = hi * 8;
  const int kbB = hi * 8;

  v8f acc0 = {}, acc1 = {};
  for (int k0 = 0; k0 < 520; k0 += 32) {
    v16h a;
#pragma unroll
    for (int h = 0; h < 16; ++h) {
      const int k = k0 + kbA + ((h < 8) ? h : h + 8);
      _Float16 v;
      if (k < 512)       v = hh[(size_t)m * 512 + k];
      else if (k < 520)  v = (_Float16)goals[(size_t)m * 8 + (k - 512)];
      else               v = (_Float16)0.f;
      a[h] = v;
    }
    v16h b0, b1;
#pragma unroll
    for (int h = 0; h < 16; ++h) {
      const int k = k0 + kbB + ((h < 8) ? h : h + 8);
      b0[h] = (k < 520) ? w[lm * 520 + k] : (_Float16)0.f;
      b1[h] = (k < 520) ? w[(16 + lm) * 520 + k] : (_Float16)0.f;
    }
    acc0 = wmma_f16(a, b0, acc0);
    acc1 = wmma_f16(a, b1, acc1);
  }
#pragma unroll
  for (int r = 0; r < 8; ++r) {
    const int mm = m0 + r + hi * 8;
    if (lm < 18)
      out[(size_t)mm * 18 + lm] = acc0[r] + bias[lm];
    if (16 + lm < 18)
      out[(size_t)mm * 18 + 16 + lm] = acc1[r] + bias[16 + lm];
  }
}

// ---------------------------------------------------------------------------
extern "C" void kernel_launch(void* const* d_in, const int* in_sizes, int n_in,
                              void* d_out, int out_size, void* d_ws,
                              size_t ws_size, hipStream_t stream) {
  (void)n_in; (void)out_size; (void)ws_size;
  const float* frames = (const float*)d_in[0];
  const float* goals  = (const float*)d_in[1];
  const float* w1 = (const float*)d_in[2];   const float* b1 = (const float*)d_in[3];
  const float* w2 = (const float*)d_in[4];   const float* b2 = (const float*)d_in[5];
  const float* w3 = (const float*)d_in[6];   const float* b3 = (const float*)d_in[7];
  const float* wt1 = (const float*)d_in[8];  const float* bt1 = (const float*)d_in[9];
  const float* wphi = (const float*)d_in[10];const float* bphi = (const float*)d_in[11];
  const float* wg = (const float*)d_in[12];  const float* bg = (const float*)d_in[13];
  const float* wp = (const float*)d_in[14];  const float* bp = (const float*)d_in[15];
  const float* wt2 = (const float*)d_in[16]; const float* bt2 = (const float*)d_in[17];
  const float* walpha = (const float*)d_in[18]; const float* balpha = (const float*)d_in[19];
  const float* wfc4 = (const float*)d_in[20];   const float* bfc4 = (const float*)d_in[21];
  const float* wfc5 = (const float*)d_in[22];   const float* bfc5 = (const float*)d_in[23];
  float* out = (float*)d_out;

  const int B = in_sizes[1] / 8;          // 1024

  // Workspace carve-up (f16 halves, 256B aligned)
  char* ws = (char*)d_ws;
  size_t off = 0;
  auto halloc = [&](size_t nhalves) -> _Float16* {
    _Float16* p = (_Float16*)(ws + off);
    off += ((nhalves * 2) + 255) & ~(size_t)255;
    return p;
  };
  _Float16* framesh = halloc((size_t)B * 4 * 84 * 84);
  _Float16* x1h     = halloc((size_t)B * 32 * 400);
  _Float16* x2h     = halloc((size_t)B * 64 * 81);
  _Float16* fmh     = halloc((size_t)B * 64 * 64);
  _Float16* aggh    = halloc((size_t)B * 64);
  _Float16* hhh     = halloc((size_t)B * 512);
  _Float16* w1h   = halloc(32 * 256);
  _Float16* w2h   = halloc(64 * 512);
  _Float16* w3h   = halloc(64 * 576);
  _Float16* wt1h  = halloc(32 * 64);
  _Float16* wphih = halloc(32 * 64);
  _Float16* wgh   = halloc(32 * 64);
  _Float16* wph   = halloc(64 * 32);
  _Float16* wt2h  = halloc(32 * 64);
  _Float16* wfc4h = halloc(512 * 64);
  _Float16* wfc5h = halloc(32 * 520);

  auto cvt = [&](_Float16* dst, const float* src, int n) {
    DA2QN_cvt<<<(n + 255) / 256, 256, 0, stream>>>(dst, src, n);
  };
  cvt(framesh, frames, B * 4 * 84 * 84);
  cvt(w1h, w1, 32 * 256);
  cvt(w2h, w2, 64 * 512);
  cvt(w3h, w3, 64 * 576);
  cvt(wt1h, wt1, 32 * 64);
  cvt(wphih, wphi, 32 * 64);
  cvt(wgh, wg, 32 * 64);
  cvt(wph, wp, 64 * 32);
  cvt(wt2h, wt2, 32 * 64);
  cvt(wfc4h, wfc4, 512 * 64);
  DA2QN_padw5<<<(32 * 520 + 255) / 256, 256, 0, stream>>>(wfc5h, wfc5);

  DA2QN_conv1<<<B * 25, 32, 0, stream>>>(framesh, w1h, b1, x1h);
  DA2QN_conv2<<<B * 6, 32, 0, stream>>>(x1h, w2h, b2, x2h);
  DA2QN_conv3<<<B * 4, 32, 0, stream>>>(x2h, w3h, b3, fmh);
  DA2QN_attn<<<B, 128, 0, stream>>>(fmh, goals, wt1h, bt1, wphih, bphi, wgh, bg,
                                    wph, bp, wt2h, bt2, walpha, balpha, aggh);
  DA2QN_fc4<<<(B / 16) * 8, 32, 0, stream>>>(aggh, wfc4h, bfc4, hhh);
  DA2QN_fc5<<<B / 16, 32, 0, stream>>>(hhh, goals, wfc5h, bfc5, out);
}
